// SiteNetAttentionBlock_2662879723992
// MI455X (gfx1250) — matrix operand order repaired and checked by reading, not compiled, not run on hardware
//
#include <hip/hip_runtime.h>
#include <hip/hip_bf16.h>

typedef __attribute__((ext_vector_type(16))) __bf16 v16bf;
typedef __attribute__((ext_vector_type(8)))  float  v8f;
typedef __attribute__((ext_vector_type(4)))  float  f32x4;

#define N_SITES 2048
#define M_NB    64
#define SD      64
#define H_HEADS 4
#define DIJE    768
#define HID     256
#define K_TOP   16
#define EPSLN   1e-5f

// LDS row strides (elements)
#define SA   776   // x_ije bf16 row stride (768 + 8 pad -> bank spread, 16B aligned)
#define SS   264   // stage f32 / act bf16 row stride (256 + 8 pad)

// LDS byte offsets (main kernel)
#define L_A       0
#define L_STAGE   (64 * SA * 2)               // 99328
#define L_ACT     (L_STAGE + 64 * SS * 4)     // 166912
#define L_LOGITS  (L_ACT + 64 * SS * 2)       // 200704
#define L_WATT    (L_LOGITS + 64 * 4 * 4)     // 201728
#define SMEM_MAIN (L_WATT + 4 * 64 * 4)       // 202752 bytes (< 320KB WGP LDS)

#define SMEM_K2   (64 * SS * 2 + 64 * SS * 4) // act + stage = 101376

// workspace byte offsets
#define W_XBF   0
#define W_W1T   (W_XBF + N_SITES * 256 * 2)
#define W_W2T   (W_W1T + 256 * DIJE * 2)
#define W_WATTT (W_W2T + 256 * 256 * 2)
#define W_WIFT  (W_WATTT + SD * DIJE * 2)
#define W_WGT   (W_WIFT + 256 * DIJE * 2)
#define W_AGG   (W_WGT + 256 * 256 * 2)

// output float offsets: (x_out, new_if, x_out)
#define O_XOUT1 0
#define O_NEWIF (N_SITES * 256)
#define O_XOUT2 (O_NEWIF + N_SITES * M_NB * 256)

// ---------------- prep kernels ----------------

__global__ void cvt_f32_to_bf16(const float* __restrict__ src, __bf16* __restrict__ dst, int n) {
    int i = blockIdx.x * blockDim.x + threadIdx.x;
    if (i < n) dst[i] = (__bf16)src[i];
}

// src is [K][C] row-major f32; dst is [C][K] row-major bf16 (weights transposed for B frags)
__global__ void transpose_to_bf16(const float* __restrict__ src, __bf16* __restrict__ dst,
                                  int K, int C) {
    int i = blockIdx.x * blockDim.x + threadIdx.x;
    if (i < K * C) {
        int k = i / C, c = i - k * C;
        dst[c * K + k] = (__bf16)src[i];
    }
}

// ---------------- WMMA GEMM: LDS A [64][sa] bf16  x  global W [C][K] bf16 -> LDS stage f32 ----
// Strip-mined: each wave owns a column strip of RG row tiles so one B fragment
// feeds RG WMMAs (4x fewer global weight loads on the 768-K GEMMs).

__device__ __forceinline__ void wmma_gemm(const __bf16* A, int sa,
                                          const __bf16* __restrict__ W, int K, int C,
                                          float* stage, int ss, int tid) {
    const int wave = tid >> 5, lane = tid & 31;
    const int hf = lane >> 4, ln = lane & 15;       // half-wave, lane-in-half
    const int CT = C >> 4;
    const int RG = (CT >= 8) ? 4 : 2;               // row tiles per strip (compile-time after inline)
    const int nstrips = (4 / RG) * CT;              // >= 8 so all waves are busy
    for (int s = wave; s < nstrips; s += 8) {
        const int ct = s % CT;
        const int rbase = (s / CT) * RG * 16;
        const int col = ct * 16 + ln;               // B: lane carries N
        v8f acc[4];
        #pragma unroll
        for (int r = 0; r < 4; ++r)
            if (r < RG) acc[r] = (v8f){0.f, 0.f, 0.f, 0.f, 0.f, 0.f, 0.f, 0.f};
        for (int k = 0; k < K; k += 32) {
            // B 32x16 frag, this code's order: lanes 0-15 K 0..15; lanes 16-31 K 16..31 (contiguous)
            const int kb = k + hf * 16;
            union { v16bf v; f32x4 f[2]; } ub;
            ub.f[0] = *(const f32x4*)(W + (long)col * K + kb);   // global_load_b128
            ub.f[1] = *(const f32x4*)(W + (long)col * K + kb + 8);
            // A 16x32 layout (matches B): lanes 0-15 K 0..15; lanes 16-31 K 16..31 (contiguous)
            const int ka = k + hf * 16;
            #pragma unroll
            for (int r = 0; r < 4; ++r) {
                if (r < RG) {
                    const int row = rbase + r * 16 + ln;         // A: lane carries M
                    union { v16bf v; f32x4 f[2]; } ua;
                    ua.f[0] = *(const f32x4*)(A + row * sa + ka);        // ds_load_b128
                    ua.f[1] = *(const f32x4*)(A + row * sa + ka + 8);
                    acc[r] = __builtin_amdgcn_wmma_f32_16x16x32_bf16(false, ua.v, false, ub.v,
                                                                     (short)0, acc[r], false, false);
                }
            }
        }
        // C layout: lanes 0-15: VGPR v -> M=v; lanes 16-31: M=v+8; N=lane&15
        const int c0 = ct * 16 + ln;
        #pragma unroll
        for (int r = 0; r < 4; ++r) {
            if (r < RG) {
                const int r0 = rbase + r * 16 + hf * 8;
                #pragma unroll
                for (int v = 0; v < 8; ++v) stage[(r0 + v) * ss + c0] = acc[r][v];
            }
        }
    }
}

// ---- fused bias + ReLU + LayerNorm over rows of stage; optional bf16-LDS / f32-global outputs ----

__device__ __forceinline__ void bias_relu_ln(float* stage, int ss, int C,
                                             const float* __restrict__ bias,
                                             const float* __restrict__ gain,
                                             const float* __restrict__ beta,
                                             int tid, __bf16* outb, int sb,
                                             float* outg, int gs, float* outg2) {
    const int wave = tid >> 5, lane = tid & 31;
    const float invC = 1.f / (float)C;
    for (int r = wave * 8; r < wave * 8 + 8; ++r) {
        float s = 0.f, s2 = 0.f;
        for (int c = lane; c < C; c += 32) {
            float v = stage[r * ss + c] + bias[c];
            v = fmaxf(v, 0.f);
            stage[r * ss + c] = v;
            s += v; s2 += v * v;
        }
        #pragma unroll
        for (int off = 16; off > 0; off >>= 1) {
            s  += __shfl_xor(s,  off, 32);
            s2 += __shfl_xor(s2, off, 32);
        }
        const float mu  = s * invC;
        const float inv = rsqrtf(s2 * invC - mu * mu + EPSLN);
        for (int c = lane; c < C; c += 32) {
            float v = (stage[r * ss + c] - mu) * inv * gain[c] + beta[c];
            if (outb)  outb[r * sb + c] = (__bf16)v;
            if (outg)  outg[(long)r * gs + c] = v;
            if (outg2) outg2[(long)r * gs + c] = v;
            if (!outb && !outg) stage[r * ss + c] = v;  // in-place f32 result
        }
    }
}

// ---------------- main kernel: one workgroup per site n ----------------

__global__ void __launch_bounds__(256)
site_attn_main(const __bf16* __restrict__ xbf,
               const float* __restrict__ iff,
               const unsigned char* __restrict__ mask,
               const int* __restrict__ ai, const int* __restrict__ aj,
               const __bf16* __restrict__ w1t, const float* __restrict__ b1,
               const float* __restrict__ g1, const float* __restrict__ be1,
               const __bf16* __restrict__ w2t, const float* __restrict__ b2,
               const float* __restrict__ g2, const float* __restrict__ be2,
               const float* __restrict__ wpre, const float* __restrict__ bpre,
               const __bf16* __restrict__ wattt, const float* __restrict__ batt,
               const float* __restrict__ gatt, const float* __restrict__ beatt,
               const __bf16* __restrict__ wift, const float* __restrict__ bif,
               const float* __restrict__ gif, const float* __restrict__ beif,
               float* __restrict__ agg_ws, float* __restrict__ out_newif) {
    extern __shared__ char smem[];
    __bf16* A     = (__bf16*)(smem + L_A);
    float*  stage = (float*) (smem + L_STAGE);
    __bf16* act   = (__bf16*)(smem + L_ACT);
    float*  lg    = (float*) (smem + L_LOGITS);
    float*  watt  = (float*) (smem + L_WATT);

    const int n = blockIdx.x;
    const int tid = threadIdx.x;
    const int wave = tid >> 5, lane = tid & 31;

    // 1) build x_ije = [x_i | x_j | if] as bf16 [64][768] in LDS
    for (int m = wave; m < M_NB; m += 8) {
        const int i = ai[n * M_NB + m];
        const int j = aj[n * M_NB + m];
        const unsigned int* xi = (const unsigned int*)(xbf + (long)i * 256);
        const unsigned int* xj = (const unsigned int*)(xbf + (long)j * 256);
        unsigned int* dst = (unsigned int*)(A + m * SA);
        for (int c = lane; c < 128; c += 32) dst[c] = xi[c];
        for (int c = lane; c < 128; c += 32) dst[128 + c] = xj[c];
        const float* f = iff + ((long)(n * M_NB + m)) * 256;
        __bf16* dA = A + m * SA + 512;
        for (int c = lane; c < 256; c += 32) dA[c] = (__bf16)f[c];
    }
    __syncthreads();

    // 2) h1 = LN(ReLU(x_ije @ w1 + b1))
    wmma_gemm(A, SA, w1t, DIJE, HID, stage, SS, tid);
    __syncthreads();
    bias_relu_ln(stage, SS, HID, b1, g1, be1, tid, act, SS, nullptr, 0, nullptr);
    __syncthreads();

    // 3) h2 = LN(ReLU(h1 @ w2 + b2))
    wmma_gemm(act, SS, w2t, HID, HID, stage, SS, tid);
    __syncthreads();
    bias_relu_ln(stage, SS, HID, b2, g2, be2, tid, act, SS, nullptr, 0, nullptr);
    __syncthreads();

    // 4) logits = h2 @ wpre + bpre, masked
    {
        const int m = tid >> 2, hh = tid & 3;
        float accv = bpre[hh];
        const __bf16* hr = act + m * SS;
        for (int k = 0; k < HID; ++k) accv += (float)hr[k] * wpre[k * 4 + hh];
        if (mask[n * M_NB + m]) accv = -__builtin_inff();
        lg[m * 4 + hh] = accv;
    }
    __syncthreads();

    // 5) top-16 softmax over M per head
    if (tid < H_HEADS) {
        const int hh = tid;
        float top[K_TOP];
        #pragma unroll
        for (int t = 0; t < K_TOP; ++t) top[t] = -__builtin_inff();
        for (int m = 0; m < M_NB; ++m) {
            float v = lg[m * 4 + hh];
            if (v > top[K_TOP - 1]) {
                top[K_TOP - 1] = v;
                for (int p = K_TOP - 1; p > 0 && top[p] > top[p - 1]; --p) {
                    float t2 = top[p]; top[p] = top[p - 1]; top[p - 1] = t2;
                }
            }
        }
        const float thr = top[K_TOP - 1], vmax = top[0];
        float sum = 0.f;
        for (int m = 0; m < M_NB; ++m) {
            float v = lg[m * 4 + hh];
            float e = (v >= thr) ? __expf(v - vmax) : 0.f;
            watt[hh * M_NB + m] = e;
            sum += e;
        }
        const float rs = 1.f / sum;
        for (int m = 0; m < M_NB; ++m) watt[hh * M_NB + m] *= rs;
    }
    __syncthreads();

    // 6) att_feats = LN(ReLU(x_ije @ watt + batt)), f32 in-place in stage
    wmma_gemm(A, SA, wattt, DIJE, SD, stage, SS, tid);
    __syncthreads();
    bias_relu_ln(stage, SS, SD, batt, gatt, beatt, tid, nullptr, 0, nullptr, 0, nullptr);
    __syncthreads();

    // 7) agg[d,h] = sum_m w_att[h,m] * att[m,d]  -> ws (flattened d*H + h)
    {
        const int d = tid & 63, hh = tid >> 6;
        float s = 0.f;
        for (int m = 0; m < M_NB; ++m) s += watt[hh * M_NB + m] * stage[m * SS + d];
        agg_ws[(long)n * 256 + d * 4 + hh] = s;
    }
    __syncthreads();

    // 8) new_if = LN(ReLU(x_ije @ wif + bif)) -> global
    wmma_gemm(A, SA, wift, DIJE, HID, stage, SS, tid);
    __syncthreads();
    bias_relu_ln(stage, SS, HID, bif, gif, beif, tid, nullptr, 0,
                 out_newif + (long)n * M_NB * 256, 256, nullptr);
}

// ---------------- kernel 2: x_out = LN(ReLU(agg @ wg + bg)), 64 sites per block ----------------

__global__ void __launch_bounds__(256)
site_xout_kernel(const float* __restrict__ agg_ws, const __bf16* __restrict__ wgt,
                 const float* __restrict__ bg, const float* __restrict__ gg,
                 const float* __restrict__ beg,
                 float* __restrict__ out1, float* __restrict__ out2) {
    extern __shared__ char smem[];
    __bf16* act   = (__bf16*)(smem);
    float*  stage = (float*)(smem + 64 * SS * 2);
    const int tid = threadIdx.x;
    const int n0 = blockIdx.x * 64;
    for (int idx = tid; idx < 64 * 256; idx += 256) {
        int r = idx >> 8, c = idx & 255;
        act[r * SS + c] = (__bf16)agg_ws[(long)(n0 + r) * 256 + c];
    }
    __syncthreads();
    wmma_gemm(act, SS, wgt, 256, 256, stage, SS, tid);
    __syncthreads();
    bias_relu_ln(stage, SS, 256, bg, gg, beg, tid, nullptr, 0,
                 out1 + (long)n0 * 256, 256, out2 + (long)n0 * 256);
}

// ---------------- host launcher ----------------

extern "C" void kernel_launch(void* const* d_in, const int* in_sizes, int n_in,
                              void* d_out, int out_size, void* d_ws, size_t ws_size,
                              hipStream_t stream) {
    (void)in_sizes; (void)n_in; (void)out_size; (void)ws_size;
    const float* x     = (const float*)d_in[0];
    const float* iff   = (const float*)d_in[1];
    const unsigned char* mask = (const unsigned char*)d_in[2];
    const int* ai      = (const int*)d_in[3];
    const int* aj      = (const int*)d_in[4];
    const float* w1    = (const float*)d_in[5];
    const float* b1    = (const float*)d_in[6];
    const float* g1    = (const float*)d_in[7];
    const float* be1   = (const float*)d_in[8];
    const float* w2    = (const float*)d_in[9];
    const float* b2    = (const float*)d_in[10];
    const float* g2    = (const float*)d_in[11];
    const float* be2   = (const float*)d_in[12];
    const float* wpre  = (const float*)d_in[13];
    const float* bpre  = (const float*)d_in[14];
    const float* watt  = (const float*)d_in[15];
    const float* batt  = (const float*)d_in[16];
    const float* gatt  = (const float*)d_in[17];
    const float* beatt = (const float*)d_in[18];
    const float* wif   = (const float*)d_in[19];
    const float* bif   = (const float*)d_in[20];
    const float* gif   = (const float*)d_in[21];
    const float* beif  = (const float*)d_in[22];
    const float* wg    = (const float*)d_in[23];
    const float* bg    = (const float*)d_in[24];
    const float* gg    = (const float*)d_in[25];
    const float* beg   = (const float*)d_in[26];

    char* ws = (char*)d_ws;
    __bf16* xbf   = (__bf16*)(ws + W_XBF);
    __bf16* w1t   = (__bf16*)(ws + W_W1T);
    __bf16* w2t   = (__bf16*)(ws + W_W2T);
    __bf16* wattt = (__bf16*)(ws + W_WATTT);
    __bf16* wift  = (__bf16*)(ws + W_WIFT);
    __bf16* wgt   = (__bf16*)(ws + W_WGT);
    float*  agg   = (float*)(ws + W_AGG);
    float*  out   = (float*)d_out;

    const int nx = N_SITES * 256;
    cvt_f32_to_bf16<<<(nx + 255) / 256, 256, 0, stream>>>(x, xbf, nx);
    transpose_to_bf16<<<(DIJE * 256 + 255) / 256, 256, 0, stream>>>(w1, w1t, DIJE, 256);
    transpose_to_bf16<<<(256 * 256 + 255) / 256, 256, 0, stream>>>(w2, w2t, 256, 256);
    transpose_to_bf16<<<(DIJE * 64 + 255) / 256, 256, 0, stream>>>(watt, wattt, DIJE, 64);
    transpose_to_bf16<<<(DIJE * 256 + 255) / 256, 256, 0, stream>>>(wif, wift, DIJE, 256);
    transpose_to_bf16<<<(256 * 256 + 255) / 256, 256, 0, stream>>>(wg, wgt, 256, 256);

    site_attn_main<<<N_SITES, 256, SMEM_MAIN, stream>>>(
        xbf, iff, mask, ai, aj,
        w1t, b1, g1, be1, w2t, b2, g2, be2, wpre, bpre,
        wattt, batt, gatt, beatt, wift, bif, gif, beif,
        agg, out + O_NEWIF);

    site_xout_kernel<<<N_SITES / 64, 256, SMEM_K2, stream>>>(
        agg, wgt, bg, gg, beg, out + O_XOUT1, out + O_XOUT2);
}
